// GraphEmbedder_old_45938970198275
// MI455X (gfx1250) — matrix operand order repaired and checked by reading, not compiled, not run on hardware
//
#include <hip/hip_runtime.h>

// ---------------- problem constants ----------------
#define NPTS   10000
#define KNN    20
#define NK     (NPTS * KNN)        // 200000
#define NEDGE  (2 * NK + NPTS)     // 410000
#define DOUTF  128                 // all layer outputs are 128 wide
#define NEG_SLOPE 0.2f
#define TILE   2048                // KNN LDS tile (3*2048*4 = 24KB)
#define MTPW   5                   // M-tiles per wave: 625 = 5 * 125 (exact)

typedef __attribute__((ext_vector_type(16))) _Float16 v16h;
typedef __attribute__((ext_vector_type(8)))  _Float16 v8h;
typedef __attribute__((ext_vector_type(8)))  float    v8f;

#if __has_builtin(__builtin_amdgcn_global_load_async_to_lds_b32) && \
    __has_builtin(__builtin_amdgcn_s_wait_asynccnt)
#define HAS_ASYNC_LDS 1
#else
#define HAS_ASYNC_LDS 0
#endif

// ---- float ordered-uint mapping for atomicMax on floats ----
__device__ __forceinline__ unsigned f2ord(float f) {
  unsigned u = __float_as_uint(f);
  return (u & 0x80000000u) ? ~u : (u | 0x80000000u);
}
__device__ __forceinline__ float ord2f(unsigned u) {
  return (u & 0x80000000u) ? __uint_as_float(u & 0x7fffffffu)
                           : __uint_as_float(~u);
}
#define ORD_NEG_INF 0x007FFFFFu   // f2ord(-inf)

// ---- decode edge e -> (src, dst, valid) from neighbor table ----
__device__ __forceinline__ void edge_decode(int e, const int* __restrict__ nbr,
                                            int& src, int& dst, bool& valid) {
  if (e < NK) {                       // (neighbor, node)
    int n = e / KNN;
    src = nbr[e]; dst = n; valid = (src != dst);
  } else if (e < 2 * NK) {            // (node, neighbor)
    int t = e - NK;
    int n = t / KNN;
    src = n; dst = nbr[t]; valid = (src != dst);
  } else {                            // self loop
    int n = e - 2 * NK;
    src = n; dst = n; valid = true;
  }
}

// ---------------- [64,N] -> [N,64] transpose of input features (f32 + f16 copies) ------
__global__ void feat_transpose(const float* __restrict__ F, float* __restrict__ X0,
                               _Float16* __restrict__ Xh) {
  int t = blockIdx.x * blockDim.x + threadIdx.x;
  if (t >= NPTS * 64) return;
  int n = t / 64, d = t % 64;
  float v = F[(size_t)d * NPTS + n];
  X0[t] = v;
  Xh[t] = (_Float16)v;
}

// ---------------- pack W [Din,128] f32 -> B-fragment-ordered f16 -----------------------
// Whb[((kt*128 + col)*32) + (half*16 + kk)] = W[(kt*32 + half*16 + kk)*128 + col]
// so a lane's whole v16h B fragment is one contiguous 32-byte run.
__global__ void w_pack(const float* __restrict__ W, _Float16* __restrict__ Whb, int Din) {
  int t = blockIdx.x * blockDim.x + threadIdx.x;
  if (t >= Din * DOUTF) return;
  int idx  = t & 31;         // half*16 + kk
  int rest = t >> 5;
  int col  = rest & 127;
  int kt   = rest >> 7;
  int k    = kt * 32 + idx;
  Whb[t] = (_Float16)W[(size_t)k * DOUTF + col];
}

// ---------------- KNN: per-thread top-20 with (async) LDS-staged coordinate tiles ------
__global__ __launch_bounds__(256) void knn_kernel(const float* __restrict__ C,
                                                  int* __restrict__ nbr) {
  __shared__ float sx[TILE], sy[TILE], sz[TILE];
  int i = blockIdx.x * blockDim.x + threadIdx.x;
  bool act = (i < NPTS);
  float xi = 0.f, yi = 0.f, zi = 0.f;
  if (act) { xi = C[i]; yi = C[NPTS + i]; zi = C[2 * NPTS + i]; }

  float bd[KNN];
  int   bi[KNN];
#pragma unroll
  for (int k = 0; k < KNN; ++k) { bd[k] = __builtin_inff(); bi[k] = 0; }

  for (int base = 0; base < NPTS; base += TILE) {
    int cnt = min(TILE, NPTS - base);
#if HAS_ASYNC_LDS
    for (int t = threadIdx.x; t < cnt; t += blockDim.x) {
      __builtin_amdgcn_global_load_async_to_lds_b32(
          (int*)(C + base + t),            (int*)(sx + t), 0, 0);
      __builtin_amdgcn_global_load_async_to_lds_b32(
          (int*)(C + NPTS + base + t),     (int*)(sy + t), 0, 0);
      __builtin_amdgcn_global_load_async_to_lds_b32(
          (int*)(C + 2 * NPTS + base + t), (int*)(sz + t), 0, 0);
    }
    __builtin_amdgcn_s_wait_asynccnt(0);
#else
    for (int t = threadIdx.x; t < cnt; t += blockDim.x) {
      sx[t] = C[base + t];
      sy[t] = C[NPTS + base + t];
      sz[t] = C[2 * NPTS + base + t];
    }
#endif
    __syncthreads();
    // prefetch next tile while we crunch this one
    int nb = base + TILE;
    if (nb + (int)threadIdx.x < NPTS) {
      int p = nb + threadIdx.x;
      __builtin_prefetch(&C[p], 0, 0);
      __builtin_prefetch(&C[NPTS + p], 0, 0);
      __builtin_prefetch(&C[2 * NPTS + p], 0, 0);
    }
    if (act) {
      for (int t = 0; t < cnt; ++t) {
        float dx = xi - sx[t], dy = yi - sy[t], dz = zi - sz[t];
        float d = dx * dx + dy * dy + dz * dz;
        if (d < bd[KNN - 1]) {                 // rare path: bubble insert
          float cd = d; int ci = base + t;
#pragma unroll
          for (int s = 0; s < KNN; ++s) {
            if (cd < bd[s]) {
              float tf = bd[s]; bd[s] = cd; cd = tf;
              int   ti = bi[s]; bi[s] = ci; ci = ti;
            }
          }
        }
      }
    }
    __syncthreads();
  }
  if (act) {
#pragma unroll
    for (int k = 0; k < KNN; ++k) nbr[i * KNN + k] = bi[k];
  }
}

// ---------------- per-layer init: segment-max = -inf, denom = 0, out = 0 ----------------
__global__ void layer_init(unsigned* __restrict__ MORD, float* __restrict__ DEN,
                           float* __restrict__ OUT) {
  int t = blockIdx.x * blockDim.x + threadIdx.x;
  if (t < NPTS) { MORD[t] = ORD_NEG_INF; DEN[t] = 0.f; }
  if (t < NPTS * DOUTF) OUT[t] = 0.f;
}

// ---------------- WMMA GEMM: H[N,128] = Xh[N,DIN] * W[DIN,128] (f16 in, f32 acc) -------
// one wave computes 5 M-tiles x 1 N-tile (80x16); 625 M-tiles = 5 * 125 -> no tail,
// branch-free straight-line body, B fragment shared across the 5 WMMAs per k-step.
template <int DIN>
__global__ __launch_bounds__(32) void gemm_wmma(const _Float16* __restrict__ Xh,
                                                const _Float16* __restrict__ Whb,
                                                float* __restrict__ H) {
  const int lane = threadIdx.x;
  const int half = lane >> 4;     // lane group 0..15 / 16..31
  const int l16  = lane & 15;
  const int nBase = blockIdx.x * 16;     // column tile (Dout = 128)
  const int mT0   = blockIdx.y * MTPW;   // first of 5 row tiles

  v8f acc[MTPW] = {};
#pragma unroll
  for (int kt = 0; kt < DIN / 32; ++kt) {
    // B: whole fragment contiguous (2 x b128)
    v16h b = *(const v16h*)(Whb + ((size_t)kt * DOUTF + nBase + l16) * 32 + half * 16);
#pragma unroll
    for (int i = 0; i < MTPW; ++i) {
      const int row = (mT0 + i) * 16 + l16;
      const _Float16* xr = Xh + (size_t)row * DIN + kt * 32 + half * 16;
      v8h lo = *(const v8h*)xr;                  // K  0.. 7 (lanes<16) / 16..23
      v8h hi = *(const v8h*)(xr + 8);            // K  8..15 (lanes<16) / 24..31
      v16h a;
#pragma unroll
      for (int j = 0; j < 8; ++j) { a[j] = lo[j]; a[8 + j] = hi[j]; }
      acc[i] = __builtin_amdgcn_wmma_f32_16x16x32_f16(
          false, a, false, b, (short)0, acc[i], false, false);
    }
  }
#pragma unroll
  for (int i = 0; i < MTPW; ++i) {
#pragma unroll
    for (int r = 0; r < 8; ++r) {
      int row = (mT0 + i) * 16 + r + (half << 3);
      H[(size_t)row * DOUTF + nBase + l16] = acc[i][r];
    }
  }
}

// ---------------- per-node attention dots: AS[n]=h[n].a_src, AD[n]=h[n].a_dst -----------
__global__ __launch_bounds__(256) void node_dots(const float* __restrict__ H,
                                                 const float* __restrict__ a_s,
                                                 const float* __restrict__ a_d,
                                                 float* __restrict__ AS,
                                                 float* __restrict__ AD) {
  int wave = (blockIdx.x * blockDim.x + threadIdx.x) >> 5;
  int lane = threadIdx.x & 31;
  if (wave >= NPTS) return;
  const float* h = H + (size_t)wave * DOUTF;
  float ss = 0.f, sd = 0.f;
#pragma unroll
  for (int f = lane; f < DOUTF; f += 32) {
    float hv = h[f];
    ss += hv * a_s[f];
    sd += hv * a_d[f];
  }
#pragma unroll
  for (int o = 16; o > 0; o >>= 1) {
    ss += __shfl_xor(ss, o, 32);
    sd += __shfl_xor(sd, o, 32);
  }
  if (lane == 0) { AS[wave] = ss; AD[wave] = sd; }
}

// ---------------- edge pass 1: logits + segment max (float atomicMax trick) ------------
__global__ void edge_logit_max(const int* __restrict__ nbr,
                               const float* __restrict__ AS,
                               const float* __restrict__ AD,
                               float* __restrict__ ELOG,
                               unsigned* __restrict__ MORD) {
  int e = blockIdx.x * blockDim.x + threadIdx.x;
  if (e >= NEDGE) return;
  int src, dst; bool valid;
  edge_decode(e, nbr, src, dst, valid);
  float logit = -__builtin_inff();
  if (valid) {
    float z = AS[src] + AD[dst];
    logit = (z > 0.f) ? z : NEG_SLOPE * z;
    atomicMax(&MORD[dst], f2ord(logit));
  }
  ELOG[e] = logit;
}

// ---------------- edge pass 2: exp(logit - max) + segment sum --------------------------
__global__ void edge_exp_sum(const int* __restrict__ nbr,
                             const unsigned* __restrict__ MORD,
                             float* __restrict__ ELOG,   // in: logit, out: exp term
                             float* __restrict__ DEN) {
  int e = blockIdx.x * blockDim.x + threadIdx.x;
  if (e >= NEDGE) return;
  int src, dst; bool valid;
  edge_decode(e, nbr, src, dst, valid);
  float m = ord2f(MORD[dst]);                 // finite: every dst has a self loop
  float ex = expf(ELOG[e] - m);               // exp(-inf) = 0 for masked edges
  ELOG[e] = ex;
  if (ex > 0.f) atomicAdd(&DEN[dst], ex);
}

// ---------------- edge pass 3: out[dst] += alpha * h[src] (128 threads per edge) -------
__global__ __launch_bounds__(128) void edge_aggregate(const int* __restrict__ nbr,
                                                      const float* __restrict__ EX,
                                                      const float* __restrict__ DEN,
                                                      const float* __restrict__ H,
                                                      float* __restrict__ OUT) {
  int e = blockIdx.x;
  int f = threadIdx.x;
  float ex = EX[e];
  if (ex == 0.f) return;
  int src, dst; bool valid;
  edge_decode(e, nbr, src, dst, valid);
  float alpha = ex / DEN[dst];
  atomicAdd(&OUT[(size_t)dst * DOUTF + f], alpha * H[(size_t)src * DOUTF + f]);
}

// ---------------- bias add -> next layer input (f32 + f16 copies) ----------------------
__global__ void bias_copy(const float* __restrict__ OUT, const float* __restrict__ b,
                          float* __restrict__ Xn, _Float16* __restrict__ Xh) {
  int t = blockIdx.x * blockDim.x + threadIdx.x;
  if (t >= NPTS * DOUTF) return;
  float v = OUT[t] + b[t & (DOUTF - 1)];
  Xn[t] = v;
  Xh[t] = (_Float16)v;
}

// ---------------- final [N,128] -> [128,N] transpose ----------------
__global__ void out_transpose(const float* __restrict__ Xf, float* __restrict__ O) {
  int t = blockIdx.x * blockDim.x + threadIdx.x;
  if (t >= NPTS * DOUTF) return;
  int d = t / NPTS, n = t % NPTS;
  O[t] = Xf[(size_t)n * DOUTF + d];
}

// =====================================================================================
extern "C" void kernel_launch(void* const* d_in, const int* in_sizes, int n_in,
                              void* d_out, int out_size, void* d_ws, size_t ws_size,
                              hipStream_t stream) {
  const float* coords = (const float*)d_in[0];
  const float* feats  = (const float*)d_in[1];
  const float* Ws[3]  = {(const float*)d_in[2],  (const float*)d_in[6],  (const float*)d_in[10]};
  const float* As[3]  = {(const float*)d_in[3],  (const float*)d_in[7],  (const float*)d_in[11]};
  const float* Ad[3]  = {(const float*)d_in[4],  (const float*)d_in[8],  (const float*)d_in[12]};
  const float* Bs[3]  = {(const float*)d_in[5],  (const float*)d_in[9],  (const float*)d_in[13]};
  const int    Dins[3] = {64, 128, 128};

  // ---- workspace carve-up (256B aligned) ----
  char* ws = (char*)d_ws;
  size_t off = 0;
  auto carve = [&](size_t bytes) -> void* {
    void* p = ws + off;
    off = (off + bytes + 255) & ~(size_t)255;
    return p;
  };
  int*       nbr  = (int*)carve((size_t)NK * sizeof(int));
  float*     xbuf = (float*)carve((size_t)NPTS * DOUTF * sizeof(float));
  float*     hbuf = (float*)carve((size_t)NPTS * DOUTF * sizeof(float));
  float*     obuf = (float*)carve((size_t)NPTS * DOUTF * sizeof(float));
  _Float16*  xh   = (_Float16*)carve((size_t)NPTS * DOUTF * sizeof(_Float16));
  _Float16*  whb  = (_Float16*)carve((size_t)DOUTF * DOUTF * sizeof(_Float16));
  float*     ASb  = (float*)carve((size_t)NPTS * sizeof(float));
  float*     ADb  = (float*)carve((size_t)NPTS * sizeof(float));
  unsigned*  MORD = (unsigned*)carve((size_t)NPTS * sizeof(unsigned));
  float*     DEN  = (float*)carve((size_t)NPTS * sizeof(float));
  float*     ELOG = (float*)carve((size_t)NEDGE * sizeof(float));
  (void)ws_size; (void)in_sizes; (void)n_in; (void)out_size;

  // 1) features [64,N] -> x0 [N,64] (f32 + f16)
  feat_transpose<<<(NPTS * 64 + 255) / 256, 256, 0, stream>>>(feats, xbuf, xh);

  // 2) KNN graph
  knn_kernel<<<(NPTS + 255) / 256, 256, 0, stream>>>(coords, nbr);

  // 3) three GAT layers
  dim3 gemm_grid(DOUTF / 16, (NPTS / 16) / MTPW);   // (8, 125), exact
  for (int l = 0; l < 3; ++l) {
    layer_init<<<(NPTS * DOUTF + 255) / 256, 256, 0, stream>>>(MORD, DEN, obuf);

    w_pack<<<(Dins[l] * DOUTF + 255) / 256, 256, 0, stream>>>(Ws[l], whb, Dins[l]);

    if (l == 0) gemm_wmma<64><<<gemm_grid, 32, 0, stream>>>(xh, whb, hbuf);
    else        gemm_wmma<128><<<gemm_grid, 32, 0, stream>>>(xh, whb, hbuf);

    node_dots<<<(NPTS * 32 + 255) / 256, 256, 0, stream>>>(hbuf, As[l], Ad[l], ASb, ADb);

    edge_logit_max<<<(NEDGE + 255) / 256, 256, 0, stream>>>(nbr, ASb, ADb, ELOG, MORD);
    edge_exp_sum  <<<(NEDGE + 255) / 256, 256, 0, stream>>>(nbr, MORD, ELOG, DEN);
    edge_aggregate<<<NEDGE, 128, 0, stream>>>(nbr, ELOG, DEN, hbuf, obuf);

    bias_copy<<<(NPTS * DOUTF + 255) / 256, 256, 0, stream>>>(obuf, Bs[l], xbuf, xh);
  }

  // 4) [N,128] -> [1,128,N]
  out_transpose<<<(NPTS * DOUTF + 255) / 256, 256, 0, stream>>>(xbuf, (float*)d_out);
}
